// SeqAttentionBlock_81793357185629
// MI455X (gfx1250) — compile-verified, matrix operand order checked by reading, not run on hardware
//
#include <hip/hip_runtime.h>
#include <hip/hip_bf16.h>

typedef __attribute__((ext_vector_type(16))) _Float16 v16h;
typedef __attribute__((ext_vector_type(8)))  float    v8f;

#define DMODEL 512
#define NHEADS 8
#define DH     64
#define T1V    192
#define BVDIM  256        // B*V = 8*32
#define BDIM   8
#define TOKENS (BVDIM * T1V)   // 49152
#define NEGINF (-1e9f)

// ---------------------------------------------------------------------------
// Fragment load: 16x32 f16 tile, row-major source with leading dim `ld`.
// A-matrix: rows r0..r0+15, K k0..k0+31.  B-matrix: pass Bt (N-major) and the
// same mapping yields lane = column N, halves = K (ISA 7.12.2 layout):
//   lane L: row = r0 + (L&15);  halves 0..7  -> K = k0 + 8*(L>>4) + 0..7
//                               halves 8..15 -> K = k0 + 8*(L>>4) + 16..23
// ---------------------------------------------------------------------------
__device__ __forceinline__ v16h ld_frag(const _Float16* __restrict__ base,
                                        int r0, int k0, int ld) {
    int lane = threadIdx.x & 31;
    const _Float16* p = base + (size_t)(r0 + (lane & 15)) * ld
                             + k0 + ((lane >> 4) << 3);
    v16h a;
#pragma unroll
    for (int i = 0; i < 8; ++i) a[i] = p[i];
#pragma unroll
    for (int i = 0; i < 8; ++i) a[8 + i] = p[16 + i];
    return a;
}

__device__ __forceinline__ v8f wmma16(v16h a, v16h b, v8f c) {
    // emits v_wmma_f32_16x16x32_f16
    return __builtin_amdgcn_wmma_f32_16x16x32_f16(false, a, false, b,
                                                  (short)0, c, false, false);
}

// ---------------------------------------------------------------------------
// 1) LayerNorm -> f16.  One wave per 512-elem row.
// ---------------------------------------------------------------------------
__global__ void ln_kernel(const float* __restrict__ x,
                          const float* __restrict__ gamma,
                          const float* __restrict__ beta,
                          _Float16* __restrict__ xn) {
    int row  = blockIdx.x * 8 + (threadIdx.x >> 5);
    int lane = threadIdx.x & 31;
    const float* xr = x + (size_t)row * DMODEL;
    float s = 0.f, ss = 0.f;
#pragma unroll
    for (int c = lane; c < DMODEL; c += 32) {
        float v = xr[c]; s += v; ss += v * v;
    }
#pragma unroll
    for (int m = 16; m >= 1; m >>= 1) {
        s  += __shfl_xor(s,  m, 32);
        ss += __shfl_xor(ss, m, 32);
    }
    float mu  = s * (1.f / DMODEL);
    float var = ss * (1.f / DMODEL) - mu * mu;
    float rs  = rsqrtf(var + 1e-5f);
    _Float16* yr = xn + (size_t)row * DMODEL;
#pragma unroll
    for (int c = lane; c < DMODEL; c += 32)
        yr[c] = (_Float16)((xr[c] - mu) * rs * gamma[c] + beta[c]);
}

// ---------------------------------------------------------------------------
// 2) Weight transpose + f16 convert: wt[n*K + k] = (f16) w[k*N + n]
// ---------------------------------------------------------------------------
__global__ void wt_kernel(const float* __restrict__ w, _Float16* __restrict__ wt,
                          int K, int N) {
    int idx = blockIdx.x * 256 + threadIdx.x;
    if (idx >= K * N) return;
    int n = idx / K, k = idx % K;
    wt[idx] = (_Float16)w[(size_t)k * N + n];
}

// ---------------------------------------------------------------------------
// 3) Mask precompute: obs_ext (BV,T1) f32, pad_ext (B,T1) f32
// ---------------------------------------------------------------------------
__global__ void mask_kernel(const int* __restrict__ obs,
                            const unsigned char* __restrict__ pad,  // numpy bool
                            float* __restrict__ o_ext,
                            float* __restrict__ p_ext) {
    int idx = blockIdx.x * 256 + threadIdx.x;
    if (idx < BVDIM * T1V) {
        int bv = idx / T1V, t = idx % T1V;
        o_ext[idx] = (t == 0) ? 1.f : (float)obs[bv * (T1V - 1) + t - 1];
    }
    if (idx < BDIM * T1V) {
        int b = idx / T1V, t = idx % T1V;
        p_ext[idx] = (t == 0) ? 1.f : (pad[b * (T1V - 1) + t - 1] ? 1.f : 0.f);
    }
}

// ---------------------------------------------------------------------------
// 4) QKV GEMM: (49152x512) @ (512x1536), 32x64 tile per wave (8 accumulators,
//    8 WMMA per 12 b128 loads per k-step).  A 64-wide N block sits inside one
//    of {q,k,v} and one head; a 32-tall M block stays inside one bv.
//    Scatter into Q/K/V (BV,H,T1,dh) f16; Q gets 1/sqrt(dh)=0.125 folded in.
// ---------------------------------------------------------------------------
__global__ void qkv_gemm(const _Float16* __restrict__ xn,
                         const _Float16* __restrict__ wt,  // (1536,512) = W^T
                         _Float16* __restrict__ Q,
                         _Float16* __restrict__ Kmat,
                         _Float16* __restrict__ V) {
    int wave = blockIdx.x * 8 + (threadIdx.x >> 5);
    const int NT = (3 * DMODEL) / 64;   // 24 col blocks
    int mt = wave / NT, nt = wave % NT;
    int m0 = mt * 32, n0 = nt * 64;
    v8f acc[2][4];
#pragma unroll
    for (int mi = 0; mi < 2; ++mi)
#pragma unroll
        for (int j = 0; j < 4; ++j) acc[mi][j] = (v8f){};

    for (int k0 = 0; k0 < DMODEL; k0 += 32) {
        v16h a0 = ld_frag(xn, m0,      k0, DMODEL);
        v16h a1 = ld_frag(xn, m0 + 16, k0, DMODEL);
        if (k0 + 32 < DMODEL) {   // hint next A k-step (global_prefetch_b8)
            __builtin_prefetch(xn + (size_t)(m0 + (threadIdx.x & 31)) * DMODEL
                                  + k0 + 32, 0, 1);
        }
#pragma unroll
        for (int j = 0; j < 4; ++j) {
            v16h b = ld_frag(wt, n0 + j * 16, k0, DMODEL);
            acc[0][j] = wmma16(a0, b, acc[0][j]);
            acc[1][j] = wmma16(a1, b, acc[1][j]);
        }
    }
    int which = n0 / DMODEL;            // 0=q 1=k 2=v (64 | 512)
    int h = (n0 % DMODEL) / DH;
    _Float16* dst = (which == 0) ? Q : (which == 1 ? Kmat : V);
    float scale = (which == 0) ? 0.125f : 1.f;
    int lane = threadIdx.x & 31;
    int colb = lane & 15;
    int rowoff = (lane >> 4) << 3;
    int bv = m0 / T1V;                  // 32 | 192 so block stays in one bv
    int tb = m0 - bv * T1V + rowoff;
    _Float16* base = dst + ((size_t)bv * NHEADS + h) * T1V * DH;
#pragma unroll
    for (int mi = 0; mi < 2; ++mi)
#pragma unroll
        for (int j = 0; j < 4; ++j)
#pragma unroll
            for (int r = 0; r < 8; ++r)
                base[(size_t)(tb + mi * 16 + r) * DH + j * 16 + colb] =
                    (_Float16)(acc[mi][j][r] * scale);
}

// ---------------------------------------------------------------------------
// 5) Attention: one block (6 waves, 192 thr) per (bv,h).  Each wave owns two
//    16-row query tiles: WMMA Q.K^T -> bias+softmax (16-lane shfl reductions)
//    -> attn (f16 via LDS re-swizzle) @ V (transposed in LDS) -> aout f16.
// ---------------------------------------------------------------------------
__global__ void __launch_bounds__(192) attn_kernel(
        const _Float16* __restrict__ Q, const _Float16* __restrict__ Kmat,
        const _Float16* __restrict__ V, const float* __restrict__ o_ext,
        const float* __restrict__ p_ext, _Float16* __restrict__ aout) {
    __shared__ _Float16 vt[DH * T1V];         // V^T: [n][k]  (24 KB)
    __shared__ _Float16 aslab[6][16 * T1V];   // per-wave attn tile (36 KB)
    __shared__ float so[T1V];
    __shared__ float sp[T1V];

    int bvh = blockIdx.x;
    int bv  = bvh / NHEADS;
    int h   = bvh % NHEADS;
    int b   = bv / 32;
    int tid = threadIdx.x, wave = tid >> 5, lane = tid & 31;

    const _Float16* Qb = Q    + (size_t)bvh * T1V * DH;
    const _Float16* Kb = Kmat + (size_t)bvh * T1V * DH;
    const _Float16* Vb = V    + (size_t)bvh * T1V * DH;

    for (int i = tid; i < T1V; i += 192) {
        so[i] = o_ext[bv * T1V + i];
        sp[i] = p_ext[b * T1V + i];
    }
    for (int i = tid; i < T1V * DH; i += 192) {
        int k = i / DH, n = i % DH;
        vt[n * T1V + k] = Vb[i];
    }
    __syncthreads();

    int colb  = lane & 15;
    int rowoff = (lane >> 4) << 3;

    for (int rt = wave; rt < T1V / 16; rt += 6) {
        int t0 = rt * 16;
        // ---- scores: 16 x 192 in 12 accumulators -------------------------
        v8f sc[12];
#pragma unroll
        for (int j = 0; j < 12; ++j) sc[j] = (v8f){};
        for (int k0 = 0; k0 < DH; k0 += 32) {
            v16h a = ld_frag(Qb, t0, k0, DH);
#pragma unroll
            for (int j = 0; j < 12; ++j) {
                v16h bf = ld_frag(Kb, j * 16, k0, DH);
                sc[j] = wmma16(a, bf, sc[j]);
            }
        }
        // ---- bias + softmax (row = t0+r+rowoff lives in one 16-lane half)
#pragma unroll
        for (int r = 0; r < 8; ++r) {
            int i = t0 + r + rowoff;
            float oi = so[i], pi = sp[i];
            float vals[12];
            float mx = -3.4e38f;
#pragma unroll
            for (int j = 0; j < 12; ++j) {
                int jj = j * 16 + colb;
                float s = sc[j][r] + oi + so[jj]
                        + ((pi * sp[jj]) != 0.f ? 0.f : NEGINF);
                vals[j] = s;
                mx = fmaxf(mx, s);
            }
#pragma unroll
            for (int m = 8; m >= 1; m >>= 1) mx = fmaxf(mx, __shfl_xor(mx, m, 32));
            float sum = 0.f;
#pragma unroll
            for (int j = 0; j < 12; ++j) { vals[j] = __expf(vals[j] - mx); sum += vals[j]; }
#pragma unroll
            for (int m = 8; m >= 1; m >>= 1) sum += __shfl_xor(sum, m, 32);
            float inv = 1.f / sum;
#pragma unroll
            for (int j = 0; j < 12; ++j)
                aslab[wave][(r + rowoff) * T1V + j * 16 + colb] =
                    (_Float16)(vals[j] * inv);
        }
        // ---- out = attn(16x192) @ V(192x64) ------------------------------
        v8f oc[4];
#pragma unroll
        for (int n = 0; n < 4; ++n) oc[n] = (v8f){};
        for (int k0 = 0; k0 < T1V; k0 += 32) {
            v16h a = ld_frag(&aslab[wave][0], 0, k0, T1V);
#pragma unroll
            for (int n = 0; n < 4; ++n) {
                v16h bf = ld_frag(vt, n * 16, k0, T1V);
                oc[n] = wmma16(a, bf, oc[n]);
            }
        }
        size_t rowbase = (size_t)bv * T1V + t0 + rowoff;
#pragma unroll
        for (int n = 0; n < 4; ++n) {
            int col = h * DH + n * 16 + colb;
#pragma unroll
            for (int r = 0; r < 8; ++r)
                aout[(rowbase + r) * DMODEL + col] = (_Float16)oc[n][r];
        }
    }
}

// ---------------------------------------------------------------------------
// 6) Output projection + bias + residual, 32x64 tile per wave, f32 out.
// ---------------------------------------------------------------------------
__global__ void proj_gemm(const _Float16* __restrict__ a,
                          const _Float16* __restrict__ wt,   // (512,512) = W^T
                          const float* __restrict__ bproj,
                          const float* __restrict__ x,
                          float* __restrict__ out) {
    int wave = blockIdx.x * 8 + (threadIdx.x >> 5);
    const int NT = DMODEL / 64;   // 8 col blocks
    int mt = wave / NT, nt = wave % NT;
    int m0 = mt * 32, n0 = nt * 64;
    v8f acc[2][4];
#pragma unroll
    for (int mi = 0; mi < 2; ++mi)
#pragma unroll
        for (int j = 0; j < 4; ++j) acc[mi][j] = (v8f){};

    for (int k0 = 0; k0 < DMODEL; k0 += 32) {
        v16h a0 = ld_frag(a, m0,      k0, DMODEL);
        v16h a1 = ld_frag(a, m0 + 16, k0, DMODEL);
        if (k0 + 32 < DMODEL) {
            __builtin_prefetch(a + (size_t)(m0 + (threadIdx.x & 31)) * DMODEL
                                 + k0 + 32, 0, 1);
        }
#pragma unroll
        for (int j = 0; j < 4; ++j) {
            v16h bf = ld_frag(wt, n0 + j * 16, k0, DMODEL);
            acc[0][j] = wmma16(a0, bf, acc[0][j]);
            acc[1][j] = wmma16(a1, bf, acc[1][j]);
        }
    }
    int lane = threadIdx.x & 31;
    int colb = lane & 15;
    int rowoff = (lane >> 4) << 3;
#pragma unroll
    for (int j = 0; j < 4; ++j) {
        int n = n0 + j * 16 + colb;
        float bb = bproj[n];
#pragma unroll
        for (int mi = 0; mi < 2; ++mi)
#pragma unroll
            for (int r = 0; r < 8; ++r) {
                size_t idx = (size_t)(m0 + mi * 16 + rowoff + r) * DMODEL + n;
                out[idx] = acc[mi][j][r] + bb + x[idx];
            }
    }
}

// ---------------------------------------------------------------------------
extern "C" void kernel_launch(void* const* d_in, const int* in_sizes, int n_in,
                              void* d_out, int out_size, void* d_ws, size_t ws_size,
                              hipStream_t stream) {
    const float* x      = (const float*)d_in[0];
    const int*   obs    = (const int*)d_in[1];
    const unsigned char* pad = (const unsigned char*)d_in[2];  // bool mask
    const float* gamma  = (const float*)d_in[3];
    const float* beta   = (const float*)d_in[4];
    const float* W_qkv  = (const float*)d_in[5];
    const float* W_proj = (const float*)d_in[6];
    const float* b_proj = (const float*)d_in[7];
    float* out = (float*)d_out;

    char* ws = (char*)d_ws;
    size_t off = 0;
    auto carve = [&](size_t bytes) {
        void* p = ws + off;
        off += (bytes + 255) & ~(size_t)255;
        return p;
    };
    _Float16* xn    = (_Float16*)carve((size_t)TOKENS * DMODEL * 2);
    _Float16* wtq   = (_Float16*)carve((size_t)3 * DMODEL * DMODEL * 2);
    _Float16* wtp   = (_Float16*)carve((size_t)DMODEL * DMODEL * 2);
    _Float16* Qbuf  = (_Float16*)carve((size_t)BVDIM * NHEADS * T1V * DH * 2);
    _Float16* Kbuf  = (_Float16*)carve((size_t)BVDIM * NHEADS * T1V * DH * 2);
    _Float16* Vbuf  = (_Float16*)carve((size_t)BVDIM * NHEADS * T1V * DH * 2);
    _Float16* abuf  = (_Float16*)carve((size_t)TOKENS * DMODEL * 2);
    float*    o_ext = (float*)carve((size_t)BVDIM * T1V * 4);
    float*    p_ext = (float*)carve((size_t)BDIM * T1V * 4);

    ln_kernel<<<TOKENS / 8, 256, 0, stream>>>(x, gamma, beta, xn);
    wt_kernel<<<(3 * DMODEL * DMODEL + 255) / 256, 256, 0, stream>>>(W_qkv, wtq, DMODEL, 3 * DMODEL);
    wt_kernel<<<(DMODEL * DMODEL + 255) / 256, 256, 0, stream>>>(W_proj, wtp, DMODEL, DMODEL);
    mask_kernel<<<(BVDIM * T1V + 255) / 256, 256, 0, stream>>>(obs, pad, o_ext, p_ext);

    // QKV: (TOKENS/32)*(1536/64) = 36864 wave-tiles, 8 waves/block
    qkv_gemm<<<(TOKENS / 32) * (3 * DMODEL / 64) / 8, 256, 0, stream>>>(xn, wtq, Qbuf, Kbuf, Vbuf);

    attn_kernel<<<BVDIM * NHEADS, 192, 0, stream>>>(Qbuf, Kbuf, Vbuf, o_ext, p_ext, abuf);

    // Proj: (TOKENS/32)*(512/64) = 12288 wave-tiles, 8 waves/block
    proj_gemm<<<(TOKENS / 32) * (DMODEL / 64) / 8, 256, 0, stream>>>(abuf, wtp, b_proj, x, out);
}
